// Quantizer_27264452395333
// MI455X (gfx1250) — matrix operand order repaired and checked by reading, not compiled, not run on hardware
//
#include <hip/hip_runtime.h>
#include <math.h>

#define N_VEC 16384
#define D_DIM 256
#define K_CB  4096
#define CHUNK 32                  // codebook entries per LDS tile
#define NCHUNK (K_CB / CHUNK)     // 128

typedef __attribute__((ext_vector_type(16))) __bf16 v16bf;
typedef __attribute__((ext_vector_type(8)))  float  v8f;

union BF16x16 {
  uint4 q[2];
  unsigned short u[16];
  v16bf v;
};

__device__ __forceinline__ unsigned short f32_to_bf16(float f) {
  unsigned int x = __float_as_uint(f);
  x += 0x7FFFu + ((x >> 16) & 1u);      // round-to-nearest-even
  return (unsigned short)(x >> 16);
}

__device__ __forceinline__ float bf16_to_f32(unsigned short h) {
  return __uint_as_float(((unsigned int)h) << 16);
}

// ---- async global->LDS copy (CDNA5), compile-safe guards ------------------
#if __has_builtin(__builtin_amdgcn_global_load_async_to_lds_b128)
#define HAVE_ASYNC_LDS 1
typedef int v4i32 __attribute__((vector_size(16)));
typedef __attribute__((address_space(1))) v4i32 g_v4i32;
typedef __attribute__((address_space(3))) v4i32 l_v4i32;
__device__ __forceinline__ void async_cp16(const void* g, void* l) {
  __builtin_amdgcn_global_load_async_to_lds_b128(
      (g_v4i32*)(uintptr_t)g,
      (l_v4i32*)(unsigned int)(uintptr_t)l, 0, 0);
}
__device__ __forceinline__ void wait_async0() {
#if __has_builtin(__builtin_amdgcn_s_wait_asynccnt)
  __builtin_amdgcn_s_wait_asynccnt(0);
#else
  asm volatile("s_wait_asynccnt 0x0" ::: "memory");
#endif
}
#else
#define HAVE_ASYNC_LDS 0
#endif

// ---------------------------------------------------------------------------
// 0) zero counts + loss accumulator in workspace
// ---------------------------------------------------------------------------
__global__ void vq_init_ws(float* wsf) {
  for (int i = threadIdx.x; i < 1 + K_CB; i += blockDim.x) wsf[i] = 0.0f;
}

// ---------------------------------------------------------------------------
// 1) one-time codebook f32 -> bf16 hi/lo split (removes VALU from hot loop)
// ---------------------------------------------------------------------------
__global__ void vq_cvt_cb(const float* __restrict__ cb,
                          unsigned short* __restrict__ cbh,
                          unsigned short* __restrict__ cbl) {
  int i = blockIdx.x * 256 + threadIdx.x;     // i < K_CB * D_DIM
  float f = cb[i];
  unsigned short h = f32_to_bf16(f);
  cbh[i] = h;
  cbl[i] = f32_to_bf16(f - bf16_to_f32(h));
}

// ---------------------------------------------------------------------------
// 2) codebook squared norms: one wave per entry
// ---------------------------------------------------------------------------
__global__ void vq_cbnorm(const float* __restrict__ cb, float* __restrict__ cbn) {
  int wave = threadIdx.x >> 5;
  int lane = threadIdx.x & 31;
  int entry = blockIdx.x * 8 + wave;
  const float* r = cb + (size_t)entry * D_DIM;
  float s = 0.0f;
#pragma unroll
  for (int j = 0; j < D_DIM / 32; ++j) {
    float v = r[lane + 32 * j];
    s += v * v;
  }
#pragma unroll
  for (int m = 16; m >= 1; m >>= 1) s += __shfl_xor(s, m);
  if (lane == 0) cbn[entry] = s;
}

// ---------------------------------------------------------------------------
// 3) main kernel: bf16 hi/lo split WMMA GEMM + fused running argmin
//    128 threads (4 waves); each wave owns 16 ze rows; double-buffered
//    async LDS staging of pre-converted codebook tiles (32 entries/tile).
// ---------------------------------------------------------------------------
__global__ __launch_bounds__(128) void vq_argmin_kernel(
    const float* __restrict__ ze,
    const unsigned short* __restrict__ cbh,
    const unsigned short* __restrict__ cbl,
    const float* __restrict__ cbn, float* counts, int* idxws,
    float* out_idx_f) {
  __shared__ __align__(16) unsigned short lds_h[2][CHUNK * D_DIM];
  __shared__ __align__(16) unsigned short lds_l[2][CHUNK * D_DIM];

  const int tid  = threadIdx.x;
  const int wave = tid >> 5;
  const int lane = tid & 31;
  const int lane16 = lane & 15;
  const int half   = lane >> 4;     // 0: lanes 0-15, 1: lanes 16-31
  const int offA = half * 16;       // A layout: lanes 0-15 K=0..15, lanes 16-31 K=16..31
  const int offB = half * 16;       // B layout: lanes 0-15 K=0..15, lanes 16-31 K=16..31

  // --- this wave's 16 ze rows resident in registers as bf16 hi/lo (A layout)
  const int row = blockIdx.x * 64 + wave * 16 + lane16;
  const float* zr = ze + (size_t)row * D_DIM;
  BF16x16 ah[8], al[8];
#pragma unroll
  for (int s = 0; s < 8; ++s) {
#pragma unroll
    for (int g = 0; g < 2; ++g) {
      int dbase = s * 32 + g * 8 + offA;
#pragma unroll
      for (int i = 0; i < 8; ++i) {
        float f = zr[dbase + i];
        unsigned short h = f32_to_bf16(f);
        ah[s].u[g * 8 + i] = h;
        al[s].u[g * 8 + i] = f32_to_bf16(f - bf16_to_f32(h));
      }
    }
  }

  float minv[8];
  int   mini[8];
#pragma unroll
  for (int v = 0; v < 8; ++v) { minv[v] = 3.4e38f; mini[v] = 0; }

  // --- stage chunk 0 into buffer 0 ---
  {
    const uint4* gh = (const uint4*)cbh;
    const uint4* gl = (const uint4*)cbl;
    uint4* sh = (uint4*)lds_h[0];
    uint4* sl = (uint4*)lds_l[0];
#pragma unroll
    for (int k = 0; k < (CHUNK * D_DIM) / (8 * 128); ++k) {   // 8 iters
      int e = tid + k * 128;
#if HAVE_ASYNC_LDS
      async_cp16(gh + e, sh + e);
      async_cp16(gl + e, sl + e);
#else
      sh[e] = gh[e];
      sl[e] = gl[e];
#endif
    }
#if HAVE_ASYNC_LDS
    wait_async0();
#endif
    __syncthreads();
  }

  const int T1 = (16 * D_DIM) / 8;   // uint4 offset of entries 16..31 in tile

  for (int c = 0; c < NCHUNK; ++c) {
    const int cur = c & 1, nxt = cur ^ 1;

    // --- issue async staging of next chunk into the other buffer ---
    if (c + 1 < NCHUNK) {
      const uint4* gh = (const uint4*)(cbh + (size_t)(c + 1) * CHUNK * D_DIM);
      const uint4* gl = (const uint4*)(cbl + (size_t)(c + 1) * CHUNK * D_DIM);
      uint4* sh = (uint4*)lds_h[nxt];
      uint4* sl = (uint4*)lds_l[nxt];
#pragma unroll
      for (int k = 0; k < (CHUNK * D_DIM) / (8 * 128); ++k) {
        int e = tid + k * 128;
#if HAVE_ASYNC_LDS
        async_cp16(gh + e, sh + e);
        async_cp16(gl + e, sl + e);
#else
        sh[e] = gh[e];
        sl[e] = gl[e];
#endif
      }
    }

    // --- compute on current buffer: two 16-col tiles, two accumulators ---
    float cn0 = cbn[c * CHUNK + lane16];
    float cn1 = cbn[c * CHUNK + 16 + lane16];
    const uint4* ph = (const uint4*)(lds_h[cur] + lane16 * D_DIM);
    const uint4* pl = (const uint4*)(lds_l[cur] + lane16 * D_DIM);

    v8f acc0 = {0.f, 0.f, 0.f, 0.f, 0.f, 0.f, 0.f, 0.f};
    v8f acc1 = {0.f, 0.f, 0.f, 0.f, 0.f, 0.f, 0.f, 0.f};
#pragma unroll
    for (int s = 0; s < 8; ++s) {
      int qi = (s * 32 + offB) >> 3;
      BF16x16 bh0, bl0, bh1, bl1;
      bh0.q[0] = ph[qi];      bh0.q[1] = ph[qi + 1];
      bl0.q[0] = pl[qi];      bl0.q[1] = pl[qi + 1];
      bh1.q[0] = ph[T1 + qi]; bh1.q[1] = ph[T1 + qi + 1];
      bl1.q[0] = pl[T1 + qi]; bl1.q[1] = pl[T1 + qi + 1];
      // split-precision dot (hi*hi + hi*lo + lo*hi), alternating accumulators
      acc0 = __builtin_amdgcn_wmma_f32_16x16x32_bf16(false, ah[s].v, false, bh0.v,
                                                     (short)0, acc0, false, false);
      acc1 = __builtin_amdgcn_wmma_f32_16x16x32_bf16(false, ah[s].v, false, bh1.v,
                                                     (short)0, acc1, false, false);
      acc0 = __builtin_amdgcn_wmma_f32_16x16x32_bf16(false, ah[s].v, false, bl0.v,
                                                     (short)0, acc0, false, false);
      acc1 = __builtin_amdgcn_wmma_f32_16x16x32_bf16(false, ah[s].v, false, bl1.v,
                                                     (short)0, acc1, false, false);
      acc0 = __builtin_amdgcn_wmma_f32_16x16x32_bf16(false, al[s].v, false, bh0.v,
                                                     (short)0, acc0, false, false);
      acc1 = __builtin_amdgcn_wmma_f32_16x16x32_bf16(false, al[s].v, false, bh1.v,
                                                     (short)0, acc1, false, false);
    }

    const int k0 = c * CHUNK + lane16;
    const int k1 = k0 + 16;
#pragma unroll
    for (int v = 0; v < 8; ++v) {
      float s0 = cn0 - 2.0f * acc0[v];   // ||e||^2 - 2*ze.e (||ze||^2 const/row)
      if (s0 < minv[v]) { minv[v] = s0; mini[v] = k0; }
      float s1 = cn1 - 2.0f * acc1[v];
      if (s1 < minv[v]) { minv[v] = s1; mini[v] = k1; }
    }

#if HAVE_ASYNC_LDS
    wait_async0();                      // next buffer's copies (mine) complete
#endif
    __syncthreads();                    // all waves: next tile ready, cur free
  }

  // --- cross-lane argmin within each 16-lane half (each half = 8 rows) ---
#pragma unroll
  for (int v = 0; v < 8; ++v) {
    float mv = minv[v];
    int   mi = mini[v];
#pragma unroll
    for (int m = 1; m < 16; m <<= 1) {
      float ov = __shfl_xor(mv, m);
      int   oi = __shfl_xor(mi, m);
      if (ov < mv || (ov == mv && oi < mi)) { mv = ov; mi = oi; }
    }
    minv[v] = mv; mini[v] = mi;
  }

  if (lane16 == 0) {
    int rbase = blockIdx.x * 64 + wave * 16 + half * 8;
#pragma unroll
    for (int v = 0; v < 8; ++v) {
      idxws[rbase + v]     = mini[v];
      out_idx_f[rbase + v] = (float)mini[v];
      atomicAdd(&counts[mini[v]], 1.0f);
    }
  }
}

// ---------------------------------------------------------------------------
// 4) gather zq = codebook[idx] and accumulate sum((zq-ze)^2)
// ---------------------------------------------------------------------------
__global__ void vq_gather_loss(const float* __restrict__ ze,
                               const float* __restrict__ cb,
                               const int* __restrict__ idxws,
                               float* __restrict__ zq_out, float* loss_acc) {
  int i = blockIdx.x * 256 + threadIdx.x;
  int r = i >> 8;
  int d = i & 255;
  int k = idxws[r];
  float q = cb[(size_t)k * D_DIM + d];
  float z = ze[i];
  zq_out[i] = q;
  float diff = q - z;
  float s = diff * diff;
#pragma unroll
  for (int m = 16; m >= 1; m >>= 1) s += __shfl_xor(s, m);
  __shared__ float wsum[8];
  if ((threadIdx.x & 31) == 0) wsum[threadIdx.x >> 5] = s;
  __syncthreads();
  if (threadIdx.x == 0) {
    float t = 0.0f;
#pragma unroll
    for (int j = 0; j < 8; ++j) t += wsum[j];
    atomicAdd(loss_acc, t);
  }
}

// ---------------------------------------------------------------------------
// 5) losses + perplexity
// ---------------------------------------------------------------------------
__global__ void vq_finalize(const float* loss_acc, const float* counts,
                            float* out3) {
  __shared__ float red[256];
  float s = 0.0f;
  for (int k = threadIdx.x; k < K_CB; k += 256) {
    float p = counts[k] * 0.1f;
    s += p * logf(p + 1e-10f);
  }
  red[threadIdx.x] = s;
  __syncthreads();
  for (int st = 128; st > 0; st >>= 1) {
    if (threadIdx.x < st) red[threadIdx.x] += red[threadIdx.x + st];
    __syncthreads();
  }
  if (threadIdx.x == 0) {
    float loss = loss_acc[0] / (float)(N_VEC * D_DIM);
    out3[0] = loss;
    out3[1] = loss;
    out3[2] = expf(-red[0]);
  }
}

extern "C" void kernel_launch(void* const* d_in, const int* in_sizes, int n_in,
                              void* d_out, int out_size, void* d_ws, size_t ws_size,
                              hipStream_t stream) {
  const float* ze = (const float*)d_in[0];        // [16384, 256] f32
  const float* cb = (const float*)d_in[1];        // [4096, 256]  f32
  float* out = (float*)d_out;
  float* out_idx = out;
  float* out_zq  = out + N_VEC;
  float* out3    = out + N_VEC + (size_t)N_VEC * D_DIM;

  // workspace: loss(1f) | counts(4096f) | cbnorm(4096f) | idx(16384 i32)
  //            | cbh (K*D u16, 16B aligned) | cbl (K*D u16)
  float* wsf      = (float*)d_ws;
  float* loss_acc = wsf;
  float* counts   = wsf + 1;
  float* cbn      = wsf + 1 + K_CB;
  int*   idxws    = (int*)(wsf + 1 + 2 * K_CB);
  size_t cbh_off  = (((size_t)(1 + 2 * K_CB) * 4 + (size_t)N_VEC * 4) + 15) & ~(size_t)15;
  unsigned short* cbh = (unsigned short*)((char*)d_ws + cbh_off);
  unsigned short* cbl = cbh + (size_t)K_CB * D_DIM;

  vq_init_ws<<<1, 256, 0, stream>>>(wsf);
  vq_cvt_cb<<<(K_CB * D_DIM) / 256, 256, 0, stream>>>(cb, cbh, cbl);
  vq_cbnorm<<<K_CB / 8, 256, 0, stream>>>(cb, cbn);
  vq_argmin_kernel<<<N_VEC / 64, 128, 0, stream>>>(ze, cbh, cbl, cbn, counts,
                                                   idxws, out_idx);
  vq_gather_loss<<<(N_VEC * D_DIM) / 256, 256, 0, stream>>>(ze, cb, idxws,
                                                            out_zq, loss_acc);
  vq_finalize<<<1, 256, 0, stream>>>(loss_acc, counts, out3);
}
